// GraphTransformerLayer_15075335209471
// MI455X (gfx1250) — matrix operand order repaired and checked by reading, not compiled, not run on hardware
//
#include <hip/hip_runtime.h>
#include <hip/hip_bf16.h>
#include <math.h>

// Problem constants (match reference)
#define B_   4
#define N_   1024
#define C_   512
#define H_   8
#define R_   7
#define E_   16384
#define HD_  64
#define M_   (B_ * N_)        // 4096 rows
#define EPS_ 1e-5f

typedef _Float16 v8h  __attribute__((ext_vector_type(8)));
typedef _Float16 v16h __attribute__((ext_vector_type(16)));
typedef float    v8f  __attribute__((ext_vector_type(8)));

union Frag16 { v16h v; v8h h[2]; };

static __device__ __forceinline__ v8f wmma_f16(v16h a, v16h b, v8f c) {
  // D = A(16x32 f16) * B(32x16 f16) + C(16x16 f32)
  return __builtin_amdgcn_wmma_f32_16x16x32_f16(false, a, false, b, (short)0, c,
                                                false, false);
}

// ---------------------------------------------------------------------------
// 1) relation map: init to -1, scatter-max of relation ids, pack to u8 (rel+1)
// ---------------------------------------------------------------------------
__global__ void k_init_rel(int* __restrict__ rel, int n) {
  int i = blockIdx.x * blockDim.x + threadIdx.x;
  if (i < n) rel[i] = -1;
}

__global__ void k_scatter_rel(const int* __restrict__ ei, const int* __restrict__ et,
                              int* __restrict__ rel) {
  int i = blockIdx.x * blockDim.x + threadIdx.x;   // [0, B*E)
  int b = i / E_, e = i - b * E_;
  int s = ei[(b * 2) * E_ + e];
  int t = ei[(b * 2 + 1) * E_ + e];
  int r = et[b * E_ + e];
  if (s >= 0 && s < N_ && t >= 0 && t < N_)
    atomicMax(&rel[b * N_ * N_ + s * N_ + t], r);
}

__global__ void k_rel8(const int* __restrict__ rel, unsigned int* __restrict__ rel8,
                       int n4) {
  int i = blockIdx.x * blockDim.x + threadIdx.x;   // packs 4 ints -> 4 bytes
  if (i >= n4) return;
  int4 v = ((const int4*)rel)[i];
  unsigned int p = ((unsigned int)(v.x + 1) & 0xff) |
                   (((unsigned int)(v.y + 1) & 0xff) << 8) |
                   (((unsigned int)(v.z + 1) & 0xff) << 16) |
                   (((unsigned int)(v.w + 1) & 0xff) << 24);
  rel8[i] = p;
}

// ---------------------------------------------------------------------------
// 2) tiled weight transpose + f32->f16:  w[K x N] -> wt[N x K]   (64x64 tiles)
// ---------------------------------------------------------------------------
__global__ void __launch_bounds__(256) k_wt(const float* __restrict__ w,
                                            _Float16* __restrict__ wt,
                                            int K, int Ncols) {
  __shared__ float tile[64][65];
  int tx = threadIdx.x & 63;
  int ty = threadIdx.x >> 6;            // 0..3
  int ntx = Ncols >> 6;
  int tk = blockIdx.x / ntx;
  int tn = blockIdx.x - tk * ntx;
  int k0 = tk * 64, n0 = tn * 64;
#pragma unroll
  for (int j = 0; j < 16; j++) {        // coalesced reads along n
    int k = ty + 4 * j;
    tile[k][tx] = w[(size_t)(k0 + k) * Ncols + n0 + tx];
  }
  __syncthreads();
#pragma unroll
  for (int j = 0; j < 16; j++) {        // coalesced writes along k
    int n = ty + 4 * j;
    wt[(size_t)(n0 + n) * K + k0 + tx] = (_Float16)tile[tx][n];
  }
}

// ---------------------------------------------------------------------------
// 3) LayerNorm (512 cols), f32 in -> f16 out. One 128-thread block per row.
// ---------------------------------------------------------------------------
__global__ void __launch_bounds__(128) k_ln(const float* __restrict__ x,
                                            const float* __restrict__ g,
                                            const float* __restrict__ b,
                                            _Float16* __restrict__ out) {
  __shared__ float red[128];
  int row = blockIdx.x, t = threadIdx.x;
  const float* xr = x + (size_t)row * C_;
  float v0[4], s = 0.f;
#pragma unroll
  for (int j = 0; j < 4; j++) { v0[j] = xr[t + 128 * j]; s += v0[j]; }
  red[t] = s; __syncthreads();
  for (int off = 64; off > 0; off >>= 1) {
    if (t < off) red[t] += red[t + off];
    __syncthreads();
  }
  float mean = red[0] * (1.0f / C_); __syncthreads();
  float vs = 0.f;
#pragma unroll
  for (int j = 0; j < 4; j++) { float d = v0[j] - mean; vs += d * d; }
  red[t] = vs; __syncthreads();
  for (int off = 64; off > 0; off >>= 1) {
    if (t < off) red[t] += red[t + off];
    __syncthreads();
  }
  float rstd = rsqrtf(red[0] * (1.0f / C_) + EPS_);
#pragma unroll
  for (int j = 0; j < 4; j++) {
    int c = t + 128 * j;
    out[(size_t)row * C_ + c] = (_Float16)((v0[j] - mean) * rstd * g[c] + b[c]);
  }
}

// ---------------------------------------------------------------------------
// 4) WMMA GEMM: out[M x N] = A[M x K](f16) * Wt[N x K]^T(f16) + bias
//    One wave per 16x64 tile. EPI: 0 = f16 store, 1 = f32 store + residual,
//    2 = exact GELU then f16 store.
// ---------------------------------------------------------------------------
template <int EPI>
__global__ void __launch_bounds__(256) k_gemm(const _Float16* __restrict__ A,
                                              const _Float16* __restrict__ Wt,
                                              const float* __restrict__ bias,
                                              const float* __restrict__ res,
                                              void* __restrict__ outp,
                                              int M, int Ncols, int K) {
  int wave = (blockIdx.x * 256 + threadIdx.x) >> 5;
  int lane = threadIdx.x & 31;
  int ntiles = Ncols >> 6;
  int mt = wave / ntiles, nt = wave - mt * ntiles;
  int m0 = mt * 16, n0 = nt * 64;
  int lh = lane & 15, lg = lane >> 4;

  const _Float16* arow = A + (size_t)(m0 + lh) * K;
  const _Float16* brow = Wt + (size_t)(n0 + lh) * K;

  v8f acc[4] = {};
  for (int k0 = 0; k0 < K; k0 += 32) {
    Frag16 a;
    a.h[0] = *(const v8h*)(arow + k0 + lg * 16);
    a.h[1] = *(const v8h*)(arow + k0 + lg * 16 + 8);
#pragma unroll
    for (int jj = 0; jj < 4; jj++) {
      Frag16 bfr;
      const _Float16* bp = brow + (size_t)(jj * 16) * K + k0 + lg * 16;
      bfr.h[0] = *(const v8h*)(bp);
      bfr.h[1] = *(const v8h*)(bp + 8);
      acc[jj] = wmma_f16(a.v, bfr.v, acc[jj]);
    }
  }

#pragma unroll
  for (int jj = 0; jj < 4; jj++) {
#pragma unroll
    for (int r = 0; r < 8; r++) {
      int n = n0 + jj * 16 + lh;
      int m = m0 + r + lg * 8;
      size_t idx = (size_t)m * Ncols + n;
      float v = acc[jj][r] + bias[n];
      if (EPI == 0) {
        ((_Float16*)outp)[idx] = (_Float16)v;
      } else if (EPI == 1) {
        ((float*)outp)[idx] = v + res[idx];
      } else {
        float gl = 0.5f * v * (1.0f + erff(v * 0.70710678118654752f));
        ((_Float16*)outp)[idx] = (_Float16)gl;
      }
    }
  }
}

// ---------------------------------------------------------------------------
// 5) per-head V transpose, LDS-tiled (64 keys x 64 d):
//    v16[(b*N+n)*C + h*64+d] -> vt[((b*H+h)*64+d)*N + n]
// ---------------------------------------------------------------------------
__global__ void __launch_bounds__(256) k_vt(const _Float16* __restrict__ v,
                                            _Float16* __restrict__ vt) {
  __shared__ _Float16 tile[64][66];
  int nt = blockIdx.x & 15;             // N/64 = 16
  int bh = blockIdx.x >> 4;             // 0..31
  int b = bh >> 3, h = bh & (H_ - 1);
  int n0 = nt * 64;
  int tx = threadIdx.x & 63, ty = threadIdx.x >> 6;
#pragma unroll
  for (int j = 0; j < 16; j++) {        // coalesced reads along d
    int n = ty + 4 * j;
    tile[n][tx] = v[(size_t)(b * N_ + n0 + n) * C_ + h * HD_ + tx];
  }
  __syncthreads();
#pragma unroll
  for (int j = 0; j < 16; j++) {        // coalesced writes along n
    int d = ty + 4 * j;
    vt[((size_t)(bh * HD_ + d)) * N_ + n0 + tx] = tile[tx][d];
  }
}

// ---------------------------------------------------------------------------
// 6) Flash attention with relational bias. One wave per (b, h, 16-query tile),
//    64 keys per iteration (4 score subtiles -> 16 WMMAs/iter).
// ---------------------------------------------------------------------------
__global__ void __launch_bounds__(256) k_attn(const _Float16* __restrict__ Q,
                                              const _Float16* __restrict__ Km,
                                              const _Float16* __restrict__ Vt,
                                              const unsigned char* __restrict__ rel8,
                                              const float* __restrict__ table,
                                              _Float16* __restrict__ O) {
  // per wave: 16x64 f16 P (1024 halves) + 16-entry f32 bias table (32 halves)
  __shared__ _Float16 smem[8 * 1056];
  int widx = threadIdx.x >> 5;
  int lane = threadIdx.x & 31;
  _Float16* pbuf = smem + widx * 1056;
  float* tbl2 = (float*)(pbuf + 1024);

  int task = blockIdx.x * 8 + widx;     // B*H*(N/16) = 2048 tasks
  int qt = task & 63;
  int bh = task >> 6;
  int h = bh & (H_ - 1), b = bh >> 3;
  int q0 = qt * 16;
  int lh = lane & 15, lg = lane >> 4;

  // branchless bias table: slot 0 == 0.0 (no edge), slot 1+r == table[r][h]
  if (lane < 16) {
    int r = lane - 1;
    float tv = (r >= 0 && r < R_) ? table[r * H_ + h] : 0.f;
    tbl2[lane] = tv;
  }

  // Q fragments (16 rows x 64 d), loaded once
  const _Float16* qrow = Q + (size_t)(b * N_ + q0 + lh) * C_ + h * HD_;
  Frag16 aq0, aq1;
  aq0.h[0] = *(const v8h*)(qrow + lg * 16);
  aq0.h[1] = *(const v8h*)(qrow + lg * 16 + 8);
  aq1.h[0] = *(const v8h*)(qrow + 32 + lg * 16);
  aq1.h[1] = *(const v8h*)(qrow + 32 + lg * 16 + 8);

  const unsigned char* relb = rel8 + (size_t)b * N_ * N_;
  v8f acc[4] = {};
  float mrun[8], lrun[8];
#pragma unroll
  for (int r = 0; r < 8; r++) { mrun[r] = -3.0e38f; lrun[r] = 0.f; }
  const float scale = 0.125f;           // 1/sqrt(64)
  const v8f vzero = {};

  asm volatile("s_wait_dscnt 0" ::: "memory");  // bias table visible

  for (int k0 = 0; k0 < N_; k0 += 64) {
    if (k0 + 64 < N_) {                 // prefetch next tiles (uniform branch)
      __builtin_prefetch(Km + (size_t)(b * N_ + k0 + 64 + lane) * C_ + h * HD_, 0, 3);
      __builtin_prefetch(Vt + ((size_t)(bh * HD_ + lane * 2)) * N_ + k0 + 64, 0, 3);
    }

    // ---- scores + scale + bias, one 16-key subtile at a time
    float p[4][8];
#pragma unroll
    for (int j = 0; j < 4; j++) {
      const _Float16* kr = Km + (size_t)(b * N_ + k0 + j * 16 + lh) * C_ + h * HD_;
      Frag16 bk;
      bk.h[0] = *(const v8h*)(kr + lg * 16);
      bk.h[1] = *(const v8h*)(kr + lg * 16 + 8);
      v8f s = wmma_f16(aq0.v, bk.v, vzero);
      bk.h[0] = *(const v8h*)(kr + 32 + lg * 16);
      bk.h[1] = *(const v8h*)(kr + 32 + lg * 16 + 8);
      s = wmma_f16(aq1.v, bk.v, s);
#pragma unroll
      for (int r = 0; r < 8; r++) {
        int q = q0 + r + lg * 8;
        unsigned int e = relb[(size_t)q * N_ + k0 + j * 16 + lh];
        p[j][r] = s[r] * scale + tbl2[e];  // branchless LDS lookup
      }
    }

    // ---- online softmax: one reduction set per row for all 64 keys
    float corr[8];
#pragma unroll
    for (int r = 0; r < 8; r++) {
      float mt = fmaxf(fmaxf(p[0][r], p[1][r]), fmaxf(p[2][r], p[3][r]));
      mt = fmaxf(mt, __shfl_xor(mt, 1, 32));
      mt = fmaxf(mt, __shfl_xor(mt, 2, 32));
      mt = fmaxf(mt, __shfl_xor(mt, 4, 32));
      mt = fmaxf(mt, __shfl_xor(mt, 8, 32));
      float mnew = fmaxf(mrun[r], mt);
      corr[r] = __expf(mrun[r] - mnew);
      mrun[r] = mnew;
      float rs = 0.f;
#pragma unroll
      for (int j = 0; j < 4; j++) {
        p[j][r] = __expf(p[j][r] - mnew);
        rs += p[j][r];
      }
      rs += __shfl_xor(rs, 1, 32);
      rs += __shfl_xor(rs, 2, 32);
      rs += __shfl_xor(rs, 4, 32);
      rs += __shfl_xor(rs, 8, 32);
      lrun[r] = lrun[r] * corr[r] + rs;
    }
#pragma unroll
    for (int jj = 0; jj < 4; jj++)
#pragma unroll
      for (int r = 0; r < 8; r++) acc[jj][r] *= corr[r];

    // ---- transpose P (C-layout -> A-layout) through per-wave LDS
#pragma unroll
    for (int r = 0; r < 8; r++) {
      int row = r + lg * 8;
#pragma unroll
      for (int j = 0; j < 4; j++)
        pbuf[row * 64 + j * 16 + lh] = (_Float16)p[j][r];
    }
    asm volatile("s_wait_dscnt 0" ::: "memory");
    Frag16 pa0, pa1;
    pa0.h[0] = *(const v8h*)(pbuf + lh * 64 + lg * 16);
    pa0.h[1] = *(const v8h*)(pbuf + lh * 64 + lg * 16 + 8);
    pa1.h[0] = *(const v8h*)(pbuf + lh * 64 + 32 + lg * 16);
    pa1.h[1] = *(const v8h*)(pbuf + lh * 64 + 32 + lg * 16 + 8);

    // ---- P @ V via pre-transposed Vt (contiguous along keys)
#pragma unroll
    for (int jj = 0; jj < 4; jj++) {
      const _Float16* vr =
          Vt + ((size_t)(bh * HD_ + jj * 16 + lh)) * N_ + k0 + lg * 16;
      Frag16 bv;
      bv.h[0] = *(const v8h*)(vr);
      bv.h[1] = *(const v8h*)(vr + 8);
      acc[jj] = wmma_f16(pa0.v, bv.v, acc[jj]);
      bv.h[0] = *(const v8h*)(vr + 32);
      bv.h[1] = *(const v8h*)(vr + 40);
      acc[jj] = wmma_f16(pa1.v, bv.v, acc[jj]);
    }
  }

  // ---- normalize and store f16 head output
#pragma unroll
  for (int jj = 0; jj < 4; jj++) {
#pragma unroll
    for (int r = 0; r < 8; r++) {
      int q = q0 + r + lg * 8;
      int d = jj * 16 + lh;
      O[(size_t)(b * N_ + q) * C_ + h * HD_ + d] = (_Float16)(acc[jj][r] / lrun[r]);
    }
  }
}

// ---------------------------------------------------------------------------
// Launch
// ---------------------------------------------------------------------------
extern "C" void kernel_launch(void* const* d_in, const int* in_sizes, int n_in,
                              void* d_out, int out_size, void* d_ws, size_t ws_size,
                              hipStream_t stream) {
  const float* x    = (const float*)d_in[0];
  const int*   ei   = (const int*)d_in[1];
  const int*   et   = (const int*)d_in[2];
  const float* q_w  = (const float*)d_in[3];
  const float* q_b  = (const float*)d_in[4];
  const float* k_w  = (const float*)d_in[5];
  const float* k_b  = (const float*)d_in[6];
  const float* v_w  = (const float*)d_in[7];
  const float* v_b  = (const float*)d_in[8];
  const float* o_w  = (const float*)d_in[9];
  const float* o_b  = (const float*)d_in[10];
  const float* tbl  = (const float*)d_in[11];
  const float* ln1g = (const float*)d_in[12];
  const float* ln1b = (const float*)d_in[13];
  const float* ln2g = (const float*)d_in[14];
  const float* ln2b = (const float*)d_in[15];
  const float* f1w  = (const float*)d_in[16];
  const float* f1b  = (const float*)d_in[17];
  const float* f2w  = (const float*)d_in[18];
  const float* f2b  = (const float*)d_in[19];
  float* out = (float*)d_out;

  char* ws = (char*)d_ws;
  size_t off = 0;
  auto alloc = [&](size_t bytes) -> void* {
    void* p = ws + off;
    off += (bytes + 255) & ~(size_t)255;
    return p;
  };
  int*           rel   = (int*)alloc((size_t)B_ * N_ * N_ * 4);
  unsigned char* rel8  = (unsigned char*)alloc((size_t)B_ * N_ * N_);
  _Float16*      h16   = (_Float16*)alloc((size_t)M_ * C_ * 2);
  _Float16*      q16   = (_Float16*)alloc((size_t)M_ * C_ * 2);
  _Float16*      k16   = (_Float16*)alloc((size_t)M_ * C_ * 2);
  _Float16*      v16   = (_Float16*)alloc((size_t)M_ * C_ * 2);
  _Float16*      vt16  = (_Float16*)alloc((size_t)M_ * C_ * 2);
  _Float16*      at16  = (_Float16*)alloc((size_t)M_ * C_ * 2);
  float*         out1  = (float*)alloc((size_t)M_ * C_ * 4);
  _Float16*      h2    = (_Float16*)alloc((size_t)M_ * C_ * 2);
  _Float16*      ffn   = (_Float16*)alloc((size_t)M_ * 4 * C_ * 2);
  _Float16*      wqt   = (_Float16*)alloc((size_t)C_ * C_ * 2);
  _Float16*      wkt   = (_Float16*)alloc((size_t)C_ * C_ * 2);
  _Float16*      wvt   = (_Float16*)alloc((size_t)C_ * C_ * 2);
  _Float16*      wot   = (_Float16*)alloc((size_t)C_ * C_ * 2);
  _Float16*      wf1t  = (_Float16*)alloc((size_t)C_ * 4 * C_ * 2);
  _Float16*      wf2t  = (_Float16*)alloc((size_t)C_ * 4 * C_ * 2);

  // relation scatter map + u8 pack
  int relN = B_ * N_ * N_;
  k_init_rel<<<relN / 256, 256, 0, stream>>>(rel, relN);
  k_scatter_rel<<<(B_ * E_) / 256, 256, 0, stream>>>(ei, et, rel);
  k_rel8<<<(relN / 4) / 256, 256, 0, stream>>>(rel, (unsigned int*)rel8, relN / 4);

  // weight convert + transpose (LDS tiled)
  k_wt<<<(C_ / 64) * (C_ / 64), 256, 0, stream>>>(q_w, wqt, C_, C_);
  k_wt<<<(C_ / 64) * (C_ / 64), 256, 0, stream>>>(k_w, wkt, C_, C_);
  k_wt<<<(C_ / 64) * (C_ / 64), 256, 0, stream>>>(v_w, wvt, C_, C_);
  k_wt<<<(C_ / 64) * (C_ / 64), 256, 0, stream>>>(o_w, wot, C_, C_);
  k_wt<<<(C_ / 64) * (4 * C_ / 64), 256, 0, stream>>>(f1w, wf1t, C_, 4 * C_);
  k_wt<<<(4 * C_ / 64) * (C_ / 64), 256, 0, stream>>>(f2w, wf2t, 4 * C_, C_);

  // LN1
  k_ln<<<M_, 128, 0, stream>>>(x, ln1g, ln1b, h16);

  // Q, K, V projections (f16 out, bias fused)
  int qkvBlocks = (M_ / 16) * (C_ / 64) / 8;             // 256
  k_gemm<0><<<qkvBlocks, 256, 0, stream>>>(h16, wqt, q_b, nullptr, q16, M_, C_, C_);
  k_gemm<0><<<qkvBlocks, 256, 0, stream>>>(h16, wkt, k_b, nullptr, k16, M_, C_, C_);
  k_gemm<0><<<qkvBlocks, 256, 0, stream>>>(h16, wvt, v_b, nullptr, v16, M_, C_, C_);

  // V transpose per head (LDS tiled)
  k_vt<<<B_ * H_ * (N_ / 64), 256, 0, stream>>>(v16, vt16);

  // attention: B*H*(N/16) = 2048 wave tasks -> 256 blocks
  k_attn<<<(B_ * H_ * (N_ / 16)) / 8, 256, 0, stream>>>(q16, k16, vt16, rel8, tbl, at16);

  // output projection + residual (f32)
  k_gemm<1><<<qkvBlocks, 256, 0, stream>>>(at16, wot, o_b, x, out1, M_, C_, C_);

  // LN2
  k_ln<<<M_, 128, 0, stream>>>(out1, ln2g, ln2b, h2);

  // FFN1 (GELU, f16 out): N=2048 -> 1024 blocks
  int f1Blocks = (M_ / 16) * ((4 * C_) / 64) / 8;        // 1024
  k_gemm<2><<<f1Blocks, 256, 0, stream>>>(h2, wf1t, f1b, nullptr, ffn, M_, 4 * C_, C_);

  // FFN2 + residual -> final output (f32)
  k_gemm<1><<<qkvBlocks, 256, 0, stream>>>(ffn, wf2t, f2b, out1, out, M_, C_, 4 * C_);
}
